// SIMCLR_Loss_73426760892667
// MI455X (gfx1250) — matrix operand order repaired and checked by reading, not compiled, not run on hardware
//
#include <hip/hip_runtime.h>
#include <cstdint>

// SimCLR NT-Xent loss, fused. N=4096, D=1024, 2N=8192, tau=0.1.
// Compute-bound (137 GFLOP vs ~32MB traffic) -> f16 WMMA (16x16x32, f32 acc).
// CDNA5 path: global_load_async_to_lds_b128 double-buffered staging (ASYNCcnt),
// 320KB-class LDS tiles, fused exp/mask epilogue (sim never materialized).

typedef __attribute__((ext_vector_type(16))) _Float16 v16h;
typedef __attribute__((ext_vector_type(8)))  _Float16 v8h;
typedef __attribute__((ext_vector_type(4)))  _Float16 v4h;
typedef __attribute__((ext_vector_type(8)))  float    v8f;

constexpr int N_ROWS = 4096;
constexpr int TWO_N  = 8192;
constexpr int D      = 1024;
constexpr float INV_TAU = 10.0f;

constexpr int BM  = 128;            // block row tile
constexpr int BN  = 128;            // block col tile
constexpr int BK  = 64;             // k tile
constexpr int LDP = BK + 8;         // padded LDS row: 72 halfs = 144B (16B mult)
constexpr int NSLICE  = 4;          // column slices (parallelism + smaller serial loop)
constexpr int NCB     = (TWO_N / BN) / NSLICE;   // 16 col blocks per slice
constexpr int SSTAGES = NCB * (D / BK);          // 256 pipeline stages

// --- CDNA5 async global->LDS helpers -------------------------------------
__device__ __forceinline__ uint32_t lds_off_u32(const void* p) {
    // LDS aperture: flat addr low 32 bits == DS byte address (ISA 10.2)
    return (uint32_t)(uintptr_t)p;
}
// copy 64 contiguous bytes: IOFFSET is added to BOTH global and LDS addresses
__device__ __forceinline__ void async_copy_64B(uint32_t lds, const _Float16* g) {
    asm volatile(
        "global_load_async_to_lds_b128 %0, %1, off\n\t"
        "global_load_async_to_lds_b128 %0, %1, off offset:16\n\t"
        "global_load_async_to_lds_b128 %0, %1, off offset:32\n\t"
        "global_load_async_to_lds_b128 %0, %1, off offset:48"
        :: "v"(lds), "v"(g) : "memory");
}
__device__ __forceinline__ void wait_async_8()  { asm volatile("s_wait_asynccnt 0x8" ::: "memory"); }
__device__ __forceinline__ void wait_async_0()  { asm volatile("s_wait_asynccnt 0x0" ::: "memory"); }

// --- Pass 0: zero scalar + per-row partial arrays ------------------------
__global__ void simclr_init(float* out, float* denomArr, float* posArr) {
    const int i = blockIdx.x * 256 + threadIdx.x;
    if (i == 0) out[0] = 0.0f;
    if (i < TWO_N) { denomArr[i] = 0.0f; posArr[i] = 0.0f; }
}

// --- Pass 1: L2-normalize rows of [z_i; z_j] -> f16 k[8192][1024] --------
__global__ __launch_bounds__(256)
void simclr_normalize(const float* __restrict__ zi, const float* __restrict__ zj,
                      _Float16* __restrict__ kh) {
    const int row = blockIdx.x;
    const int t   = threadIdx.x;
    const float* src = (row < N_ROWS) ? (zi + (size_t)row * D)
                                      : (zj + (size_t)(row - N_ROWS) * D);
    const float4 v4 = ((const float4*)src)[t];
    float ss = v4.x * v4.x + v4.y * v4.y + v4.z * v4.z + v4.w * v4.w;
    #pragma unroll
    for (int off = 16; off > 0; off >>= 1) ss += __shfl_down(ss, off, 32);
    __shared__ float red[8];
    if ((t & 31) == 0) red[t >> 5] = ss;
    __syncthreads();
    float tot = 0.0f;
    #pragma unroll
    for (int i = 0; i < 8; ++i) tot += red[i];
    const float inv = rsqrtf(tot);
    v4h h;
    h[0] = (_Float16)(v4.x * inv);
    h[1] = (_Float16)(v4.y * inv);
    h[2] = (_Float16)(v4.z * inv);
    h[3] = (_Float16)(v4.w * inv);
    *(v4h*)(kh + (size_t)row * D + (size_t)t * 4) = h;
}

// --- Pass 2: fused sim = k@k^T + exp/mask epilogue -----------------------
// grid: (64 row blocks, 4 col slices); block: 256 threads = 8 waves (4x2),
// wave tile 32x64 (2 M-subtiles x 4 N-subtiles), double-buffered async LDS.
__global__ __launch_bounds__(256)
void simclr_main(const _Float16* __restrict__ kh,
                 float* __restrict__ denomArr, float* __restrict__ posArr) {
    __shared__ _Float16 As[2][BM][LDP];
    __shared__ _Float16 Bs[2][BN][LDP];
    __shared__ float sdenom[BM];
    __shared__ float spos[BM];

    const int tid  = threadIdx.x;
    const int wave = tid >> 5;
    const int lane = tid & 31;
    const int l16  = lane & 15;
    const int hi   = lane >> 4;
    const int rowBase   = blockIdx.x * BM;
    const int sliceBase = blockIdx.y * NCB;

    const int m0 = (wave >> 1) * 32;   // 0,32,64,96
    const int n0 = (wave & 1) * 64;    // 0,64

    if (tid < BM) { sdenom[tid] = 0.0f; spos[tid] = 0.0f; }

    // cooperative staging: 2 threads per row, 32 halfs (64B) each
    const int ldRow = tid >> 1;          // 0..127
    const int ldCol = (tid & 1) * 32;    // 0,32

    auto issue = [&](int s) {
        const int cb  = s >> 4;
        const int kb  = s & 15;
        const int buf = s & 1;
        const int k0  = kb * BK;
        const int colBase = (sliceBase + cb) * BN;
        const _Float16* ga = kh + (size_t)(rowBase + ldRow) * D + k0 + ldCol;
        async_copy_64B(lds_off_u32(&As[buf][ldRow][ldCol]), ga);
        const _Float16* gb = kh + (size_t)(colBase + ldRow) * D + k0 + ldCol;
        async_copy_64B(lds_off_u32(&Bs[buf][ldRow][ldCol]), gb);
    };

    float denom[2][8];
    float posv[2][8];
    #pragma unroll
    for (int r = 0; r < 2; ++r)
        #pragma unroll
        for (int v = 0; v < 8; ++v) { denom[r][v] = 0.0f; posv[r][v] = 0.0f; }

    v8f acc[2][4];

    issue(0);
    issue(1);

    for (int s = 0; s < SSTAGES; ++s) {
        const int cb  = s >> 4;
        const int kb  = s & 15;
        const int buf = s & 1;

        if (s < SSTAGES - 1) wait_async_8();   // stage s complete, s+1 in flight
        else                 wait_async_0();
        __syncthreads();                        // tile visible to all waves

        if (kb == 0) {
            const v8f z = {};
            #pragma unroll
            for (int r = 0; r < 2; ++r)
                #pragma unroll
                for (int j = 0; j < 4; ++j) acc[r][j] = z;
        }

        #pragma unroll
        for (int kc = 0; kc < BK; kc += 32) {
            // A fragments (16x32 f16): lane l16 = row, hi selects K-half
            v16h a0, a1;
            {
                const _Float16* p = &As[buf][m0 + l16][kc + hi * 8];
                v8h lo = *(const v8h*)p;
                v8h up = *(const v8h*)(p + 16);
                #pragma unroll
                for (int i = 0; i < 8; ++i) { a0[i] = lo[i]; a0[i + 8] = up[i]; }
                p = &As[buf][m0 + 16 + l16][kc + hi * 8];
                lo = *(const v8h*)p;
                up = *(const v8h*)(p + 16);
                #pragma unroll
                for (int i = 0; i < 8; ++i) { a1[i] = lo[i]; a1[i + 8] = up[i]; }
            }
            #pragma unroll
            for (int j = 0; j < 4; ++j) {
                // B fragment (32x16 f16): lane l16 = col (row of k), hi = K-half
                v16h b;
                {
                    const _Float16* q = &Bs[buf][n0 + j * 16 + l16][kc + hi * 8];
                    v8h blo = *(const v8h*)q;
                    v8h bhi = *(const v8h*)(q + 16);
                    #pragma unroll
                    for (int i = 0; i < 8; ++i) { b[i] = blo[i]; b[i + 8] = bhi[i]; }
                }
                acc[0][j] = __builtin_amdgcn_wmma_f32_16x16x32_f16(
                    false, a0, false, b, (short)0, acc[0][j], false, false);
                acc[1][j] = __builtin_amdgcn_wmma_f32_16x16x32_f16(
                    false, a1, false, b, (short)0, acc[1][j], false, false);
            }
        }

        if (kb == 15) {
            // epilogue: C layout -> lane l16 = N col, slot v = M row (+8 if hi)
            const int colBase = (sliceBase + cb) * BN;
            #pragma unroll
            for (int r = 0; r < 2; ++r)
                #pragma unroll
                for (int j = 0; j < 4; ++j) {
                    const int gcol = colBase + n0 + j * 16 + l16;
                    #pragma unroll
                    for (int v = 0; v < 8; ++v) {
                        const int grow = rowBase + m0 + r * 16 + v + hi * 8;
                        const float sv = acc[r][j][v];
                        float e = __expf(sv * INV_TAU);
                        if (gcol == grow) e = 0.0f;               // mask diagonal
                        denom[r][v] += e;
                        if (gcol == ((grow + N_ROWS) & (TWO_N - 1))) posv[r][v] += sv;
                    }
                }
        }

        __syncthreads();                        // all waves done reading buf
        if (s + 2 < SSTAGES) issue(s + 2);      // overwrite buf for stage s+2
    }

    // reduce 16 lanes per row (C-frag lane groups), merge 2 col-waves in LDS
    #pragma unroll
    for (int r = 0; r < 2; ++r)
        #pragma unroll
        for (int v = 0; v < 8; ++v) {
            float d = denom[r][v];
            float p = posv[r][v];
            #pragma unroll
            for (int off = 8; off > 0; off >>= 1) {
                d += __shfl_down(d, off, 16);
                p += __shfl_down(p, off, 16);
            }
            if (l16 == 0) {
                const int lr = m0 + r * 16 + v + hi * 8;
                atomicAdd(&sdenom[lr], d);
                atomicAdd(&spos[lr], p);
            }
        }
    __syncthreads();

    if (tid < BM) {
        atomicAdd(&denomArr[rowBase + tid], sdenom[tid]);
        atomicAdd(&posArr[rowBase + tid], spos[tid]);
    }
}

// --- Pass 3: per-row log + global mean -----------------------------------
__global__ __launch_bounds__(256)
void simclr_finalize(const float* __restrict__ denomArr, const float* __restrict__ posArr,
                     float* __restrict__ out) {
    const int r = blockIdx.x * 256 + threadIdx.x;
    float c = (__logf(denomArr[r]) - posArr[r] * INV_TAU) * (1.0f / (float)TWO_N);
    #pragma unroll
    for (int off = 16; off > 0; off >>= 1) c += __shfl_down(c, off, 32);
    if ((threadIdx.x & 31) == 0) atomicAdd(out, c);
}

extern "C" void kernel_launch(void* const* d_in, const int* in_sizes, int n_in,
                              void* d_out, int out_size, void* d_ws, size_t ws_size,
                              hipStream_t stream) {
    const float* zi = (const float*)d_in[0];   // [4096][1024] f32
    const float* zj = (const float*)d_in[1];   // [4096][1024] f32
    float* out = (float*)d_out;                // scalar f32

    _Float16* kh  = (_Float16*)d_ws;                              // 16 MB
    float* denomArr = (float*)((char*)d_ws + (size_t)TWO_N * D * sizeof(_Float16));
    float* posArr   = denomArr + TWO_N;

    simclr_init<<<TWO_N / 256, 256, 0, stream>>>(out, denomArr, posArr);
    simclr_normalize<<<TWO_N, 256, 0, stream>>>(zi, zj, kh);
    simclr_main<<<dim3(TWO_N / BM, NSLICE), 256, 0, stream>>>(kh, denomArr, posArr);
    simclr_finalize<<<TWO_N / 256, 256, 0, stream>>>(denomArr, posArr, out);
}
